// SAGEConvCuGraph_70574902608298
// MI455X (gfx1250) — matrix operand order repaired and checked by reading, not compiled, not run on hardware
//
#include <hip/hip_runtime.h>

typedef __attribute__((ext_vector_type(4)))  _Float16 v4h;
typedef __attribute__((ext_vector_type(8)))  _Float16 v8h;
typedef __attribute__((ext_vector_type(16))) _Float16 v16h;
typedef __attribute__((ext_vector_type(8)))  float    v8f;

#define N_NODES 100000
#define N_EDGES 1600000
#define F_IN    64
#define F_OUT   64
#define K_DIM   128          // 2*F_IN
#define M_TILES (N_NODES / 16)   // 6250, exact

// ---- workspace layout (≈26 MB total) ----
// [0, 25.6e6)        agg  : N_NODES x 64 f32   (later reused in-place as h: N_NODES x 128 f16)
// [25.6e6, 26.0e6)   deg  : N_NODES f32
// [26.0e6, +16KB)    Wh   : 64 x 128 f16 (same row-major layout as W)
#define AGG_OFF 0
#define DEG_OFF 25600000
#define WH_OFF  26000000

// ---------- phase 0: zero agg+deg (contiguous 6.5M floats) ----------
__global__ void zero_ws_kernel(float* __restrict__ p, int n4) {
    int i = blockIdx.x * blockDim.x + threadIdx.x;
    if (i < n4) ((float4*)p)[i] = make_float4(0.f, 0.f, 0.f, 0.f);
}

// ---------- phase 0b: W f32 -> f16, same [64,128] row-major layout ----------
__global__ void cvt_w_kernel(const float* __restrict__ W, _Float16* __restrict__ Wh) {
    int i = blockIdx.x * blockDim.x + threadIdx.x;
    if (i < F_OUT * K_DIM) Wh[i] = (_Float16)W[i];
}

// ---------- phase 1: scatter-add x[src] into agg[dst], count deg ----------
// 16 threads per edge, float4 payload per thread -> coalesced 256B per edge.
// x and agg both fit in the 192MB L2; atomics resolve in L2 atomic units.
__global__ void scatter_kernel(const float* __restrict__ x,
                               const long long* __restrict__ ei,
                               float* __restrict__ agg,
                               float* __restrict__ deg) {
    int tid = blockIdx.x * blockDim.x + threadIdx.x;   // 25.6M < 2^31
    int e = tid >> 4;
    int g = tid & 15;
    if (e >= N_EDGES) return;
    long long src = ei[e];
    long long dst = ei[N_EDGES + e];
    const float4 v = *(const float4*)(x + src * 64 + g * 4);
    float* a = agg + dst * 64 + g * 4;
    atomicAdd(a + 0, v.x);
    atomicAdd(a + 1, v.y);
    atomicAdd(a + 2, v.z);
    atomicAdd(a + 3, v.w);
    if (g == 0) atomicAdd(deg + dst, 1.0f);
}

// ---------- phase 2: agg /= max(deg,1); pack h = [agg | x] as f16 IN PLACE over agg ----------
// Block = 256 threads = 16 nodes x 16 threads. Two-phase (read-all, barrier, write-all)
// because the f16 output aliases the f32 agg bytes of the same node.
__global__ void norm_pack_kernel(const float* __restrict__ x,
                                 float* __restrict__ agg,
                                 const float* __restrict__ deg) {
    int node = blockIdx.x * 16 + (threadIdx.x >> 4);
    int g = threadIdx.x & 15;
    float4 a  = *(const float4*)(agg + (size_t)node * 64 + g * 4);
    float4 xv = *(const float4*)(x   + (size_t)node * 64 + g * 4);
    float inv = 1.0f / fmaxf(deg[node], 1.0f);
    __syncthreads();   // all reads of this block's node bytes complete before aliasing writes
    _Float16* h = (_Float16*)agg;
    v4h ha = { (_Float16)(a.x * inv), (_Float16)(a.y * inv),
               (_Float16)(a.z * inv), (_Float16)(a.w * inv) };
    v4h hx = { (_Float16)xv.x, (_Float16)xv.y, (_Float16)xv.z, (_Float16)xv.w };
    *(v4h*)(h + (size_t)node * K_DIM + g * 4)          = ha;
    *(v4h*)(h + (size_t)node * K_DIM + F_IN + g * 4)   = hx;
}

// ---------- phase 3: out = h @ W^T + b via v_wmma_f32_16x16x32_f16 ----------
// One wave per 16-row M-tile; 4 N-tiles of 16 cols; K=128 in 4 chained WMMAs.
__global__ void __launch_bounds__(256) sage_gemm_kernel(
    const _Float16* __restrict__ h, const _Float16* __restrict__ Wh,
    const float* __restrict__ bias, float* __restrict__ out) {
    const int lane = threadIdx.x & 31;
    const int wave = threadIdx.x >> 5;
    const int mtile = blockIdx.x * 8 + wave;
    if (mtile >= M_TILES) return;              // wave-uniform: EXEC stays all-1s for WMMA

    const int  l15  = lane & 15;
    const bool hi   = lane >= 16;
    const size_t arow = (size_t)(mtile * 16 + l15) * K_DIM;  // A row for this lane
    const int aoff = hi ? 16 : 0;              // A: lanes>=16 hold K+16..K+31
    const int boff = hi ? 16 : 0;              // B: lanes>=16 hold K+16..K+31

    // C/D: all 8 elements of a lane share column n*16+l15 -> preload bias into acc
    v8f acc[4];
#pragma unroll
    for (int n = 0; n < 4; ++n) {
        float bv = bias[n * 16 + l15];
#pragma unroll
        for (int v = 0; v < 8; ++v) acc[n][v] = bv;
    }

#pragma unroll
    for (int kk = 0; kk < 4; ++kk) {
        const int k0 = kk * 32;
        // A fragment: VGPR0-3 = K[k0+aoff .. +7], VGPR4-7 = K[k0+aoff+8 .. +15]
        v8h alo = *(const v8h*)(h + arow + k0 + aoff);
        v8h ahi = *(const v8h*)(h + arow + k0 + aoff + 8);
        v16h afrag;
#pragma unroll
        for (int i = 0; i < 8; ++i) { afrag[i] = alo[i]; afrag[8 + i] = ahi[i]; }
#pragma unroll
        for (int n = 0; n < 4; ++n) {
            // B fragment: 16 consecutive K values of W row (= output column), 32B load
            v16h bfrag = *(const v16h*)(Wh + (size_t)(n * 16 + l15) * K_DIM + k0 + boff);
            acc[n] = __builtin_amdgcn_wmma_f32_16x16x32_f16(
                false, afrag, false, bfrag, (short)0, acc[n], false, false);
        }
    }

    // D layout: lanes<16 rows 0..7, lanes>=16 rows 8..15; column fixed per lane
    const int rbase = mtile * 16 + (hi ? 8 : 0);
#pragma unroll
    for (int n = 0; n < 4; ++n) {
        const int col = n * 16 + l15;
#pragma unroll
        for (int v = 0; v < 8; ++v)
            out[(size_t)(rbase + v) * F_OUT + col] = acc[n][v];
    }
}

extern "C" void kernel_launch(void* const* d_in, const int* in_sizes, int n_in,
                              void* d_out, int out_size, void* d_ws, size_t ws_size,
                              hipStream_t stream) {
    const float*     x  = (const float*)d_in[0];
    const long long* ei = (const long long*)d_in[1];   // [2, E] int64
    // d_in[2] = num_nodes scalar (compile-time constant here)
    const float*     W  = (const float*)d_in[3];
    const float*     b  = (const float*)d_in[4];
    float*           out = (float*)d_out;

    char* ws = (char*)d_ws;
    float*    agg = (float*)(ws + AGG_OFF);
    float*    deg = (float*)(ws + DEG_OFF);
    _Float16* Wh  = (_Float16*)(ws + WH_OFF);

    // agg + deg are contiguous: 6.5M floats = 1.625M float4s
    zero_ws_kernel<<<(1625000 + 255) / 256, 256, 0, stream>>>(agg, 1625000);
    cvt_w_kernel<<<(F_OUT * K_DIM + 255) / 256, 256, 0, stream>>>(W, Wh);
    scatter_kernel<<<(N_EDGES * 16) / 256, 256, 0, stream>>>(x, ei, agg, deg);
    norm_pack_kernel<<<N_NODES / 16, 256, 0, stream>>>(x, agg, deg);
    sage_gemm_kernel<<<(M_TILES + 7) / 8, 256, 0, stream>>>(
        (const _Float16*)agg, Wh, b, out);
}
